// WanEncoderAttentionBlock_16965120820016
// MI455X (gfx1250) — matrix operand order repaired and checked by reading, not compiled, not run on hardware
//
#include <hip/hip_runtime.h>

typedef unsigned short ushort_t;
typedef __attribute__((ext_vector_type(16))) __bf16 v16bf;
typedef __attribute__((ext_vector_type(8)))  float  v8f;

#define S_TOK 3872   // 8*22*22, multiple of 32
#define C_DIM 1536
#define NHEAD 12
#define DHEAD 128
#define EPS   1e-6f
#define ATT_SCALE 0.08838834764831845f   // 1/sqrt(128)
#define LOG_THETA 9.210340371976184f     // ln(10000)

// ---------------- helpers ----------------

__device__ __forceinline__ ushort_t f2bf(float f) {
    unsigned u = __builtin_bit_cast(unsigned, f);
    u += 0x7FFFu + ((u >> 16) & 1u);     // round-to-nearest-even
    return (ushort_t)(u >> 16);
}

union FragBF { v16bf v; uint4 q[2]; };

// A-matrix 16x32 bf16 fragment: lane<16 row=lane, K {0..7,16..23}; lane>=16 K {8..15,24..31}
__device__ __forceinline__ v16bf load_a_bf16(const ushort_t* base, int ldk,
                                             int row, int k0, int half) {
    const ushort_t* p = base + (size_t)row * ldk + k0 + (half ? 8 : 0);
    FragBF f;
    f.q[0] = *(const uint4*)(p);
    f.q[1] = *(const uint4*)(p + 16);
    return f.v;
}

// B-matrix 32x16 bf16 fragment: lane = column; lane<16 K {0..7,16..23}; lane>=16 K {8..15,24..31}
__device__ __forceinline__ v16bf load_b_bf16(const ushort_t* base, int ldk,
                                             int col, int k0, int half) {
    const ushort_t* p = base + (size_t)col * ldk + k0 + (half ? 8 : 0);
    FragBF f;
    f.q[0] = *(const uint4*)(p);
    f.q[1] = *(const uint4*)(p + 16);
    return f.v;
}

__device__ __forceinline__ v8f wmma_bf16(v16bf a, v16bf b, v8f c) {
    return __builtin_amdgcn_wmma_f32_16x16x32_bf16(false, a, false, b,
                                                   (short)0, c, false, false);
}

__device__ __forceinline__ float redmax16(float v) {
    v = fmaxf(v, __shfl_xor(v, 1, 32));
    v = fmaxf(v, __shfl_xor(v, 2, 32));
    v = fmaxf(v, __shfl_xor(v, 4, 32));
    v = fmaxf(v, __shfl_xor(v, 8, 32));
    return v;
}
__device__ __forceinline__ float redsum16(float v) {
    v += __shfl_xor(v, 1, 32);
    v += __shfl_xor(v, 2, 32);
    v += __shfl_xor(v, 4, 32);
    v += __shfl_xor(v, 8, 32);
    return v;
}

// ---------------- kernel 0: weights f32 -> bf16 ----------------

__global__ __launch_bounds__(256)
void k_wconv(const float* wq, const float* wk, const float* wv, const float* wo,
             ushort_t* oq, ushort_t* ok, ushort_t* ov, ushort_t* oo) {
    size_t i = (size_t)blockIdx.x * 256 + threadIdx.x;
    const float* src; ushort_t* dst;
    switch (blockIdx.y) {
        case 0: src = wq; dst = oq; break;
        case 1: src = wk; dst = ok; break;
        case 2: src = wv; dst = ov; break;
        default: src = wo; dst = oo; break;
    }
    dst[i] = f2bf(src[i]);
}

// ---------------- kernel 1: transpose + layernorm -> xn bf16 (S x C) ----------------

__global__ __launch_bounds__(256)
void k_ln(const float* __restrict__ x, ushort_t* __restrict__ xn) {
    const int s = blockIdx.x, tid = threadIdx.x;
    __shared__ float r1[256], r2[256];
    float a = 0.f, b = 0.f;
    #pragma unroll
    for (int i = 0; i < 6; ++i) {
        int c = tid + i * 256;
        float v = x[(size_t)c * S_TOK + s];
        a += v; b += v * v;
    }
    r1[tid] = a; r2[tid] = b;
    __syncthreads();
    for (int off = 128; off > 0; off >>= 1) {
        if (tid < off) { r1[tid] += r1[tid + off]; r2[tid] += r2[tid + off]; }
        __syncthreads();
    }
    const float mean = r1[0] * (1.f / C_DIM);
    const float var  = r2[0] * (1.f / C_DIM) - mean * mean;
    const float rs   = rsqrtf(var + EPS);
    #pragma unroll
    for (int i = 0; i < 6; ++i) {
        int c = tid + i * 256;
        xn[(size_t)s * C_DIM + c] = f2bf((x[(size_t)c * S_TOK + s] - mean) * rs);
    }
}

// ---------------- kernel 2: QKV projection GEMM (bf16 WMMA, f32 out) ----------------
// C[m][n] = sum_k xn[m][k] * W[n][k] + b[n];  grid (C/64, ceil(S/128), 3), 8 waves/WG

__global__ __launch_bounds__(256)
void k_qkv(const ushort_t* __restrict__ xn,
           const ushort_t* wq, const ushort_t* wk, const ushort_t* wv,
           const float* bq, const float* bk, const float* bv,
           float* outq, float* outk, float* outv) {
    const ushort_t* w; const float* bias; float* out;
    if      (blockIdx.z == 0) { w = wq; bias = bq; out = outq; }
    else if (blockIdx.z == 1) { w = wk; bias = bk; out = outk; }
    else                      { w = wv; bias = bv; out = outv; }

    const int tid = threadIdx.x, lane = tid & 31, half = lane >> 4, cl = lane & 15;
    const int waveId = tid >> 5;
    const int m_base = blockIdx.y * 128 + (waveId & 3) * 32;
    const int n_base = blockIdx.x * 64  + (waveId >> 2) * 32;

    const int rowA0 = min(m_base + cl,      S_TOK - 1);
    const int rowA1 = min(m_base + 16 + cl, S_TOK - 1);
    const int colB0 = n_base + cl;
    const int colB1 = colB0 + 16;

    v8f c00 = {}, c01 = {}, c10 = {}, c11 = {};
    for (int k0 = 0; k0 < C_DIM; k0 += 32) {
        __builtin_prefetch(w + (size_t)colB0 * C_DIM + k0 + 64, 0, 0);
        v16bf a0 = load_a_bf16(xn, C_DIM, rowA0, k0, half);
        v16bf a1 = load_a_bf16(xn, C_DIM, rowA1, k0, half);
        v16bf b0 = load_b_bf16(w,  C_DIM, colB0, k0, half);
        v16bf b1 = load_b_bf16(w,  C_DIM, colB1, k0, half);
        c00 = wmma_bf16(a0, b0, c00);
        c01 = wmma_bf16(a0, b1, c01);
        c10 = wmma_bf16(a1, b0, c10);
        c11 = wmma_bf16(a1, b1, c11);
    }
    const float bias0 = bias[colB0], bias1 = bias[colB1];
    #pragma unroll
    for (int j = 0; j < 8; ++j) {
        int r = m_base + j + (half ? 8 : 0);
        if (r < S_TOK) {
            out[(size_t)r * C_DIM + colB0] = c00[j] + bias0;
            out[(size_t)r * C_DIM + colB1] = c01[j] + bias1;
        }
        r += 16;
        if (r < S_TOK) {
            out[(size_t)r * C_DIM + colB0] = c10[j] + bias0;
            out[(size_t)r * C_DIM + colB1] = c11[j] + bias1;
        }
    }
}

// ---------------- kernel 3: RMSNorm + 3D RoPE on q,k -> bf16 ----------------

__global__ __launch_bounds__(256)
void k_rmsrope(const float* __restrict__ qr_, const float* __restrict__ kr_,
               const float* gq, const float* gk,
               ushort_t* __restrict__ qbf, ushort_t* __restrict__ kbf) {
    const int s = blockIdx.x, tid = threadIdx.x;
    __shared__ float r1[256], r2[256];
    float aq = 0.f, ak = 0.f;
    #pragma unroll
    for (int i = 0; i < 6; ++i) {
        int c = tid + i * 256;
        float q = qr_[(size_t)s * C_DIM + c]; aq += q * q;
        float k = kr_[(size_t)s * C_DIM + c]; ak += k * k;
    }
    r1[tid] = aq; r2[tid] = ak;
    __syncthreads();
    for (int off = 128; off > 0; off >>= 1) {
        if (tid < off) { r1[tid] += r1[tid + off]; r2[tid] += r2[tid + off]; }
        __syncthreads();
    }
    const float rq = rsqrtf(r1[0] * (1.f / C_DIM) + EPS);
    const float rk = rsqrtf(r2[0] * (1.f / C_DIM) + EPS);

    const int tpos = s / 484;          // T index (H*W = 484)
    const int hpos = (s / 22) % 22;    // H index
    const int wpos = s % 22;           // W index

    #pragma unroll
    for (int i = 0; i < 3; ++i) {      // 768 pairs total
        int p  = tid + i * 256;
        int h  = p >> 6;               // head
        int pp = p & 63;               // pair index within head (half = 64)
        float ang;
        if      (pp < 22) ang = (float)tpos * __expf(-(float)pp        * (LOG_THETA / 22.f));
        else if (pp < 43) ang = (float)hpos * __expf(-(float)(pp - 22) * (LOG_THETA / 21.f));
        else              ang = (float)wpos * __expf(-(float)(pp - 43) * (LOG_THETA / 21.f));
        float sn, cs;
        __sincosf(ang, &sn, &cs);
        const size_t c0 = (size_t)s * C_DIM + h * DHEAD + 2 * pp;
        {
            float xr = qr_[c0]     * rq * gq[h * DHEAD + 2 * pp];
            float xi = qr_[c0 + 1] * rq * gq[h * DHEAD + 2 * pp + 1];
            qbf[c0]     = f2bf(xr * cs - xi * sn);
            qbf[c0 + 1] = f2bf(xr * sn + xi * cs);
        }
        {
            float xr = kr_[c0]     * rk * gk[h * DHEAD + 2 * pp];
            float xi = kr_[c0 + 1] * rk * gk[h * DHEAD + 2 * pp + 1];
            kbf[c0]     = f2bf(xr * cs - xi * sn);
            kbf[c0 + 1] = f2bf(xr * sn + xi * cs);
        }
    }
}

// ---------------- kernel 4: V (S x C, f32) -> V^T (C x S, bf16) ----------------

__global__ __launch_bounds__(256)
void k_vtrans(const float* __restrict__ v, ushort_t* __restrict__ vT) {
    __shared__ float t[16][17];
    const int c0 = blockIdx.x * 16, s0 = blockIdx.y * 16;
    const int tx = threadIdx.x, ty = threadIdx.y;
    t[ty][tx] = v[(size_t)(s0 + ty) * C_DIM + c0 + tx];
    __syncthreads();
    vT[(size_t)(c0 + ty) * S_TOK + s0 + tx] = f2bf(t[tx][ty]);
}

// ---------------- kernel 5: flash attention (bf16 WMMA, online softmax) ----------------
// grid (61, 12), 128 threads = 4 waves; each wave owns 16 query rows of one head.

__global__ __launch_bounds__(128)
void k_attn(const ushort_t* __restrict__ qbf, const ushort_t* __restrict__ kbf,
            const ushort_t* __restrict__ vT,  ushort_t* __restrict__ obf) {
    __shared__ ushort_t plds[4 * 16 * 32];          // one 16x32 P tile per wave
    const int tid = threadIdx.x, lane = tid & 31, half = lane >> 4, cl = lane & 15;
    const int waveId = tid >> 5;
    const int sblock = blockIdx.x * 4 + waveId;     // 16-row query tile index
    if (sblock >= S_TOK / 16) return;               // wave-uniform exit (EXEC stays full)
    const int h = blockIdx.y;

    ushort_t* my = plds + waveId * 512;

    // preload Q fragments for the 16 rows (d = 128 -> 4 K-steps of 32)
    v16bf qa[4];
    {
        const int rowq = sblock * 16 + cl;
        #pragma unroll
        for (int kd = 0; kd < 4; ++kd)
            qa[kd] = load_a_bf16(qbf + h * DHEAD, C_DIM, rowq, kd * 32, half);
    }

    v8f acc[8] = {};                                // o tile: 16 rows x 128 dims
    float m[8], l[8];
    #pragma unroll
    for (int j = 0; j < 8; ++j) { m[j] = -1e30f; l[j] = 0.f; }

    for (int sk0 = 0; sk0 < S_TOK; sk0 += 32) {
        // ---- scores: two 16x16 tiles over 32 keys
        v8f s0 = {}, s1 = {};
        #pragma unroll
        for (int kd = 0; kd < 4; ++kd) {
            v16bf b0 = load_b_bf16(kbf + h * DHEAD, C_DIM, sk0 + cl,      kd * 32, half);
            v16bf b1 = load_b_bf16(kbf + h * DHEAD, C_DIM, sk0 + 16 + cl, kd * 32, half);
            s0 = wmma_bf16(qa[kd], b0, s0);
            s1 = wmma_bf16(qa[kd], b1, s1);
        }
        // ---- online softmax update (row r = j + 8*half, lanes replicate row stats)
        float p0[8], p1[8];
        #pragma unroll
        for (int j = 0; j < 8; ++j) {
            float a = s0[j] * ATT_SCALE, b = s1[j] * ATT_SCALE;
            float mx = redmax16(fmaxf(a, b));
            float mn = fmaxf(m[j], mx);
            float alpha = __expf(m[j] - mn);
            m[j] = mn;
            p0[j] = __expf(a - mn);
            p1[j] = __expf(b - mn);
            l[j] = l[j] * alpha + redsum16(p0[j] + p1[j]);
            #pragma unroll
            for (int db = 0; db < 8; ++db) acc[db][j] *= alpha;
        }
        // ---- P: C-layout -> LDS (16x32 row-major) -> A-fragment
        #pragma unroll
        for (int j = 0; j < 8; ++j) {
            int row = j + (half ? 8 : 0);
            my[row * 32 + cl]      = f2bf(p0[j]);
            my[row * 32 + 16 + cl] = f2bf(p1[j]);
        }
        __asm__ volatile("s_wait_dscnt 0" ::: "memory");   // wave-local LDS ordering
        v16bf pa;
        {
            const ushort_t* p = my + cl * 32 + (half ? 8 : 0);
            FragBF f;
            f.q[0] = *(const uint4*)(p);
            f.q[1] = *(const uint4*)(p + 16);
            pa = f.v;
        }
        // ---- o += P @ V  (B = V^T rows: contiguous in key index)
        #pragma unroll
        for (int db = 0; db < 8; ++db) {
            v16bf vb = load_b_bf16(vT + (size_t)h * DHEAD * S_TOK,
                                   S_TOK, db * 16 + cl, sk0, half);
            acc[db] = wmma_bf16(pa, vb, acc[db]);
        }
    }

    // ---- epilogue: o /= l, store bf16 (S x C)
    float inv[8];
    #pragma unroll
    for (int j = 0; j < 8; ++j) inv[j] = 1.f / l[j];
    #pragma unroll
    for (int db = 0; db < 8; ++db) {
        #pragma unroll
        for (int j = 0; j < 8; ++j) {
            int s = sblock * 16 + j + (half ? 8 : 0);
            int c = h * DHEAD + db * 16 + cl;
            obf[(size_t)s * C_DIM + c] = f2bf(acc[db][j] * inv[j]);
        }
    }
}

// ---------------- kernel 6: output GEMM + residual, write (C, S) f32 ----------------

__global__ __launch_bounds__(256)
void k_out(const ushort_t* __restrict__ obf, const ushort_t* __restrict__ wo,
           const float* __restrict__ bo, const float* __restrict__ x,
           float* __restrict__ y) {
    const int tid = threadIdx.x, lane = tid & 31, half = lane >> 4, cl = lane & 15;
    const int waveId = tid >> 5;
    const int m_base = blockIdx.y * 128 + (waveId & 3) * 32;
    const int n_base = blockIdx.x * 64  + (waveId >> 2) * 32;

    const int rowA0 = min(m_base + cl,      S_TOK - 1);
    const int rowA1 = min(m_base + 16 + cl, S_TOK - 1);
    const int colB0 = n_base + cl;
    const int colB1 = colB0 + 16;

    v8f c00 = {}, c01 = {}, c10 = {}, c11 = {};
    for (int k0 = 0; k0 < C_DIM; k0 += 32) {
        v16bf a0 = load_a_bf16(obf, C_DIM, rowA0, k0, half);
        v16bf a1 = load_a_bf16(obf, C_DIM, rowA1, k0, half);
        v16bf b0 = load_b_bf16(wo,  C_DIM, colB0, k0, half);
        v16bf b1 = load_b_bf16(wo,  C_DIM, colB1, k0, half);
        c00 = wmma_bf16(a0, b0, c00);
        c01 = wmma_bf16(a0, b1, c01);
        c10 = wmma_bf16(a1, b0, c10);
        c11 = wmma_bf16(a1, b1, c11);
    }
    const float bias0 = bo[colB0], bias1 = bo[colB1];
    #pragma unroll
    for (int j = 0; j < 8; ++j) {
        int r = m_base + j + (half ? 8 : 0);
        if (r < S_TOK) {
            y[(size_t)colB0 * S_TOK + r] = c00[j] + bias0 + x[(size_t)colB0 * S_TOK + r];
            y[(size_t)colB1 * S_TOK + r] = c01[j] + bias1 + x[(size_t)colB1 * S_TOK + r];
        }
        r += 16;
        if (r < S_TOK) {
            y[(size_t)colB0 * S_TOK + r] = c10[j] + bias0 + x[(size_t)colB0 * S_TOK + r];
            y[(size_t)colB1 * S_TOK + r] = c11[j] + bias1 + x[(size_t)colB1 * S_TOK + r];
        }
    }
}

// ---------------- host launcher ----------------

extern "C" void kernel_launch(void* const* d_in, const int* in_sizes, int n_in,
                              void* d_out, int out_size, void* d_ws, size_t ws_size,
                              hipStream_t stream) {
    const float* x  = (const float*)d_in[0];
    const float* Wq = (const float*)d_in[1];
    const float* bq = (const float*)d_in[2];
    const float* Wk = (const float*)d_in[3];
    const float* bk = (const float*)d_in[4];
    const float* Wv = (const float*)d_in[5];
    const float* bv = (const float*)d_in[6];
    const float* Wo = (const float*)d_in[7];
    const float* bo = (const float*)d_in[8];
    const float* gq = (const float*)d_in[9];
    const float* gk = (const float*)d_in[10];
    float* y = (float*)d_out;

    const size_t SC2 = (size_t)S_TOK * C_DIM * 2;
    const size_t CC2 = (size_t)C_DIM * C_DIM * 2;
    const size_t SC4 = (size_t)S_TOK * C_DIM * 4;

    char* w = (char*)d_ws;
    ushort_t* xn_bf  = (ushort_t*)w; w += SC2;
    ushort_t* wq_bf  = (ushort_t*)w; w += CC2;
    ushort_t* wk_bf  = (ushort_t*)w; w += CC2;
    ushort_t* wv_bf  = (ushort_t*)w; w += CC2;
    ushort_t* wo_bf  = (ushort_t*)w; w += CC2;
    float*    q_raw  = (float*)w;    w += SC4;
    float*    k_raw  = (float*)w;    w += SC4;
    float*    v_raw  = (float*)w;    w += SC4;
    ushort_t* q_bf   = (ushort_t*)w; w += SC2;
    ushort_t* k_bf   = (ushort_t*)w; w += SC2;
    ushort_t* vT_bf  = (ushort_t*)w; w += SC2;
    ushort_t* attn_bf= (ushort_t*)w; w += SC2;

    k_wconv<<<dim3((C_DIM * C_DIM) / 256, 4), 256, 0, stream>>>(
        Wq, Wk, Wv, Wo, wq_bf, wk_bf, wv_bf, wo_bf);

    k_ln<<<S_TOK, 256, 0, stream>>>(x, xn_bf);

    k_qkv<<<dim3(C_DIM / 64, (S_TOK + 127) / 128, 3), 256, 0, stream>>>(
        xn_bf, wq_bf, wk_bf, wv_bf, bq, bk, bv, q_raw, k_raw, v_raw);

    k_rmsrope<<<S_TOK, 256, 0, stream>>>(q_raw, k_raw, gq, gk, q_bf, k_bf);

    k_vtrans<<<dim3(C_DIM / 16, S_TOK / 16), dim3(16, 16), 0, stream>>>(v_raw, vT_bf);

    k_attn<<<dim3((S_TOK / 16 + 3) / 4, NHEAD), 128, 0, stream>>>(
        q_bf, k_bf, vT_bf, attn_bf);

    k_out<<<dim3(C_DIM / 64, (S_TOK + 127) / 128), 256, 0, stream>>>(
        attn_bf, wo_bf, bo, x, y);
}
